// TtCodeGenAttention_38568806318568
// MI455X (gfx1250) — matrix operand order repaired and checked by reading, not compiled, not run on hardware
//
#include <hip/hip_runtime.h>
#include <hip/hip_fp16.h>
#include <math.h>

// ---------------------------------------------------------------------------
// CodeGen attention for MI455X (gfx1250, wave32, WMMA f16 -> f32 accum)
// B=2, S=2048, E=2048, H=16, D=128, ROTARY=64, MP_NUM=4, causal
// GEMMs: async global->LDS double-buffered staging (ASYNCcnt) + WMMA
// ---------------------------------------------------------------------------

typedef __attribute__((ext_vector_type(16))) _Float16 v16h;
typedef __attribute__((ext_vector_type(8)))  _Float16 v8h;
typedef __attribute__((ext_vector_type(8)))  float    v8f;
typedef __attribute__((ext_vector_type(4)))  int      v4i;

#define DEVI __device__ __forceinline__

// ---- async global -> LDS copy (CDNA5), with safe fallback -----------------
// Builtin signature (from hipcc diagnostic): (addrspace(1) v4i* src,
// addrspace(3) v4i* dst, imm offset, imm cpol)

typedef __attribute__((address_space(1))) v4i GV4;
typedef __attribute__((address_space(3))) v4i LV4;

DEVI void cp_async_b128(_Float16* ldsDst, const _Float16* gSrc) {
#if __has_builtin(__builtin_amdgcn_global_load_async_to_lds_b128)
  __builtin_amdgcn_global_load_async_to_lds_b128((GV4*)gSrc, (LV4*)ldsDst,
                                                 0, 0);
#else
  *(v8h*)ldsDst = *(const v8h*)gSrc;   // synchronous fallback
#endif
}

DEVI void wait_async0() {
#if __has_builtin(__builtin_amdgcn_s_wait_asynccnt)
  __builtin_amdgcn_s_wait_asynccnt(0);
#endif
}
DEVI void wait_async8() {
#if __has_builtin(__builtin_amdgcn_s_wait_asynccnt)
  __builtin_amdgcn_s_wait_asynccnt(8);
#else
  // fallback copies are synchronous; nothing outstanding
#endif
}

// ---- WMMA fragment helpers (wave32, contiguous K order: lane half h holds K = 16h..16h+15) ----------

// A-matrix 16x32 f16, source row-major [16 rows, ld stride], K contiguous.
// lane<16 : row=lane,    K = 0..15
// lane>=16: row=lane-16, K = 16..31
DEVI v16h load_frag_a(const _Float16* base, int ld) {
  const int lane = threadIdx.x & 31;
  const int row  = lane & 15;
  const int koff = (lane < 16) ? 0 : 16;
  const _Float16* p = base + (size_t)row * ld + koff;
  v8h lo = *(const v8h*)(p);
  v8h hi = *(const v8h*)(p + 8);
  v16h r;
#pragma unroll
  for (int i = 0; i < 8; ++i) { r[i] = lo[i]; r[8 + i] = hi[i]; }
  return r;
}

// B-matrix 32x16 f16 from Bsrc[n*ld + k] (N rows in memory, K contiguous).
// lane<16 : col=lane,    K = 0..15 ; lane>=16: col=lane-16, K = 16..31
DEVI v16h load_frag_b(const _Float16* base, int ld) {
  const int lane = threadIdx.x & 31;
  const int col  = lane & 15;
  const int koff = (lane < 16) ? 0 : 16;
  return *(const v16h*)(base + (size_t)col * ld + koff);
}

DEVI v8f wmma_f16(v16h a, v16h b, v8f c) {
  return __builtin_amdgcn_wmma_f32_16x16x32_f16(false, a, false, b,
                                                (short)0, c, false, false);
}

// ---- fp32 -> fp16 conversion ----------------------------------------------

__global__ void cvt_f32_f16(const float* __restrict__ in,
                            _Float16* __restrict__ out, size_t n) {
  size_t i      = (size_t)blockIdx.x * blockDim.x + threadIdx.x;
  size_t stride = (size_t)gridDim.x * blockDim.x;
  for (; i < n; i += stride) out[i] = (_Float16)in[i];
}

// ---- GEMM: C[M,N] = A[M,K] * B[N,K]^T  (both f16, fp32 accumulate) --------
// Block = 256 threads = 8 waves; block tile 128x128; K-panel 64, double-
// buffered async staging in LDS (2 x (128x64 A + 128x64 B) = 64 KB).
// Wave tile 64x32 (2x4 wave grid), fragments read from LDS.

template <bool STORE_F16>
__global__ __launch_bounds__(256) void gemm_tn(const _Float16* __restrict__ A,
                                               const _Float16* __restrict__ B,
                                               void* __restrict__ C,
                                               int M, int N, int K) {
  __shared__ __align__(32) _Float16 ldsA[2][128][64];
  __shared__ __align__(32) _Float16 ldsB[2][128][64];

  const int tid  = threadIdx.x;
  const int wave = tid >> 5;
  const int lane = tid & 31;
  const int wm = wave >> 2;                       // 0..1
  const int wn = wave & 3;                        // 0..3
  const int rowBase = blockIdx.y * 128;
  const int colBase = blockIdx.x * 128;

  // Stage one 128x64 A panel + 128x64 B panel: 1024 b128 segments per panel
  // -> 8 async copies per thread (4 A + 4 B).
  auto stage = [&](int buf, int kk) {
#pragma unroll
    for (int i = 0; i < 4; ++i) {
      const int idx = tid + i * 256;
      const int row = idx >> 3;
      const int sg  = (idx & 7) * 8;
      cp_async_b128(&ldsA[buf][row][sg],
                    A + (size_t)(rowBase + row) * K + kk + sg);
      cp_async_b128(&ldsB[buf][row][sg],
                    B + (size_t)(colBase + row) * K + kk + sg);
    }
  };

  v8f acc[4][2];
  v8f z = {};
#pragma unroll
  for (int i = 0; i < 4; ++i)
#pragma unroll
    for (int j = 0; j < 2; ++j) acc[i][j] = z;

  stage(0, 0);
  int buf = 0;
  for (int kk = 0; kk < K; kk += 64) {
    const bool hasNext = (kk + 64) < K;
    if (hasNext) stage(buf ^ 1, kk + 64);   // overlap DMA with compute
    if (hasNext) wait_async8();             // previous stage (8 ops) done
    else         wait_async0();
    __syncthreads();                        // all waves' panels visible

#pragma unroll
    for (int ks = 0; ks < 64; ks += 32) {
      v16h a[4];
#pragma unroll
      for (int i = 0; i < 4; ++i)
        a[i] = load_frag_a(&ldsA[buf][wm * 64 + 16 * i][ks], 64);
      v16h b[2];
#pragma unroll
      for (int j = 0; j < 2; ++j)
        b[j] = load_frag_b(&ldsB[buf][wn * 32 + 16 * j][ks], 64);
#pragma unroll
      for (int i = 0; i < 4; ++i)
#pragma unroll
        for (int j = 0; j < 2; ++j)
          acc[i][j] = wmma_f16(a[i], b[j], acc[i][j]);
    }
    __syncthreads();                        // done reading buf before reuse
    buf ^= 1;
  }

  const int rOff = (lane < 16) ? 0 : 8;
  const int cOff = lane & 15;
  const int wrow = rowBase + wm * 64;
  const int wcol = colBase + wn * 32;
#pragma unroll
  for (int i = 0; i < 4; ++i)
#pragma unroll
    for (int j = 0; j < 2; ++j)
#pragma unroll
      for (int r = 0; r < 8; ++r) {
        size_t idx = (size_t)(wrow + 16 * i + rOff + r) * N +
                     (wcol + 16 * j + cOff);
        if (STORE_F16) ((_Float16*)C)[idx] = (_Float16)acc[i][j][r];
        else           ((float*)C)[idx]    = acc[i][j][r];
      }
}

// ---- RoPE + split qkv -> Q[B,H,S,D], K[B,H,S,D], V^T[B,H,D,S] -------------
// qkv layout per row: mp-group of 1536 = [q:512 | v:512 | k:512], head=mp*4+hh
// 1/sqrt(128) score scale folded into Q.

__global__ void rope_split(const _Float16* __restrict__ qkv,
                           _Float16* __restrict__ Q,
                           _Float16* __restrict__ Kk,
                           _Float16* __restrict__ Vt) {
  size_t idx = (size_t)blockIdx.x * blockDim.x + threadIdx.x;
  if (idx >= (size_t)2 * 16 * 2048 * 64) return;   // one thread per dim-pair
  const int dp = (int)(idx & 63);
  size_t t = idx >> 6;
  const int s = (int)(t & 2047); t >>= 11;
  const int h = (int)(t & 15);
  const int b = (int)(t >> 4);
  const int mp = h >> 2, hh = h & 3;

  const _Float16* row =
      qkv + ((size_t)(b * 2048 + s)) * 6144 + mp * 1536 + hh * 128;
  const int d0 = dp * 2, d1 = d0 + 1;

  float qe = (float)row[d0],        qo = (float)row[d1];
  float ve = (float)row[512 + d0],  vo = (float)row[512 + d1];
  float ke = (float)row[1024 + d0], ko = (float)row[1024 + d1];

  if (d0 < 64) {  // rotary part: inv_freq = 10000^(-d0/64), pair rotation
    const float freq = (float)s * __powf(10000.f, -(float)d0 / 64.f);
    float sn, cs;
    __sincosf(freq, &sn, &cs);
    float qe2 = qe * cs - qo * sn, qo2 = qo * cs + qe * sn;
    float ke2 = ke * cs - ko * sn, ko2 = ko * cs + ke * sn;
    qe = qe2; qo = qo2; ke = ke2; ko = ko2;
  }
  const float qscale = 0.08838834764831845f;  // 1/sqrt(128)
  qe *= qscale; qo *= qscale;

  const size_t bh = (size_t)(b * 16 + h);
  _Float16* qp = Q  + (bh * 2048 + s) * 128;
  _Float16* kp = Kk + (bh * 2048 + s) * 128;
  qp[d0] = (_Float16)qe; qp[d1] = (_Float16)qo;
  kp[d0] = (_Float16)ke; kp[d1] = (_Float16)ko;
  _Float16* vp = Vt + bh * 128 * 2048 + s;
  vp[(size_t)d0 * 2048] = (_Float16)ve;
  vp[(size_t)d1 * 2048] = (_Float16)vo;
}

// ---- Flash attention (causal), per wave: 16 query rows x full D=128 -------
// Block = 8 waves = 128 query rows. Online softmax; P staged via LDS to
// convert WMMA C-layout -> A-layout for the PV matmul. K/V are L2-resident
// (48 MB < 192 MB L2); prefetch next key block.

__global__ __launch_bounds__(256) void flash_attn(const _Float16* __restrict__ Q,
                                                  const _Float16* __restrict__ Km,
                                                  const _Float16* __restrict__ Vt,
                                                  _Float16* __restrict__ Ctx) {
  __shared__ __align__(32) _Float16 ldsP[8][16][32];

  const int wave = threadIdx.x >> 5;
  const int lane = threadIdx.x & 31;
  const int bh = blockIdx.y;                       // b*16 + h
  const int b  = bh >> 4;
  const int h  = bh & 15;
  const int qr0  = blockIdx.x * 128 + wave * 16;
  const int rOff = (lane < 16) ? 0 : 8;
  const int col  = lane & 15;

  const _Float16* Qp = Q  + ((size_t)bh * 2048) * 128;
  const _Float16* Kp = Km + ((size_t)bh * 2048) * 128;
  const _Float16* Vp = Vt + ((size_t)bh * 128) * 2048;

  // Q fragments for all four 32-wide chunks of D, loaded once.
  v16h qf[4];
#pragma unroll
  for (int dk = 0; dk < 4; ++dk)
    qf[dk] = load_frag_a(Qp + (size_t)qr0 * 128 + dk * 32, 128);

  v8f o[8];
  v8f z = {};
#pragma unroll
  for (int t = 0; t < 8; ++t) o[t] = z;

  float mrow[8], lrow[8];
#pragma unroll
  for (int r = 0; r < 8; ++r) { mrow[r] = -INFINITY; lrow[r] = 0.f; }

  const int nkb = (qr0 + 16 + 31) >> 5;            // causal: keys <= qr0+15
  for (int kb = 0; kb < nkb; ++kb) {
    const int k0 = kb * 32;
    if (kb + 1 < nkb) {  // prefetch next key block rows
      __builtin_prefetch(Kp + (size_t)(k0 + 32) * 128 + (lane << 2), 0, 1);
    }

    // S = Q K^T for 16 queries x 32 keys
    v8f sacc[2];
    sacc[0] = z; sacc[1] = z;
#pragma unroll
    for (int dk = 0; dk < 4; ++dk) {
#pragma unroll
      for (int j = 0; j < 2; ++j) {
        v16h bk = load_frag_b(Kp + (size_t)(k0 + 16 * j) * 128 + dk * 32, 128);
        sacc[j] = wmma_f16(qf[dk], bk, sacc[j]);
      }
    }

    // causal mask + online softmax (per-row stats; rows live in 16-lane halves)
    float crr[8];
#pragma unroll
    for (int r = 0; r < 8; ++r) {
      const int qrow = qr0 + rOff + r;
      float s0 = (k0 + col      <= qrow) ? sacc[0][r] : -INFINITY;
      float s1 = (k0 + 16 + col <= qrow) ? sacc[1][r] : -INFINITY;
      float tm = fmaxf(s0, s1);
#pragma unroll
      for (int off = 8; off >= 1; off >>= 1)
        tm = fmaxf(tm, __shfl_xor(tm, off, 32));
      const float mnew = fmaxf(mrow[r], tm);
      const float c  = __expf(mrow[r] - mnew);
      const float p0 = __expf(s0 - mnew);
      const float p1 = __expf(s1 - mnew);
      float rs = p0 + p1;
#pragma unroll
      for (int off = 8; off >= 1; off >>= 1) rs += __shfl_xor(rs, off, 32);
      lrow[r] = lrow[r] * c + rs;
      mrow[r] = mnew;
      crr[r]  = c;
      ldsP[wave][rOff + r][col]      = (_Float16)p0;
      ldsP[wave][rOff + r][col + 16] = (_Float16)p1;
    }
#pragma unroll
    for (int t = 0; t < 8; ++t)
#pragma unroll
      for (int r = 0; r < 8; ++r) o[t][r] *= crr[r];

    // O += P (16x32) * V (32x128), V^T layout makes B-fragments K-contiguous
    v16h pa = load_frag_a(&ldsP[wave][0][0], 32);
#pragma unroll
    for (int t = 0; t < 8; ++t) {
      v16h bv = load_frag_b(Vp + (size_t)(t * 16) * 2048 + k0, 2048);
      o[t] = wmma_f16(pa, bv, o[t]);
    }
  }

  // epilogue: divide by softmax sum, write ctx[b, s, h*128 + d] as f16
#pragma unroll
  for (int r = 0; r < 8; ++r) {
    const float inv = 1.f / lrow[r];
    const int srow = qr0 + rOff + r;
    _Float16* outp = Ctx + ((size_t)(b * 2048 + srow)) * 2048 + h * 128;
#pragma unroll
    for (int t = 0; t < 8; ++t)
      outp[t * 16 + col] = (_Float16)(o[t][r] * inv);
  }
}

// ---------------------------------------------------------------------------

extern "C" void kernel_launch(void* const* d_in, const int* in_sizes, int n_in,
                              void* d_out, int out_size, void* d_ws, size_t ws_size,
                              hipStream_t stream) {
  const float* hs   = (const float*)d_in[0];   // [2,2048,2048]
  const float* wqkv = (const float*)d_in[1];   // [6144,2048]
  const float* wout = (const float*)d_in[2];   // [2048,2048]

  char* ws = (char*)d_ws;
  _Float16* h16    = (_Float16*)(ws + 0);          // 16 MB
  _Float16* wqkv16 = (_Float16*)(ws + 16777216);   // 24 MB
  _Float16* wout16 = (_Float16*)(ws + 41943040);   //  8 MB
  _Float16* qkv16  = (_Float16*)(ws + 50331648);   // 48 MB
  _Float16* q16    = (_Float16*)(ws + 100663296);  // 16 MB
  _Float16* k16    = (_Float16*)(ws + 117440512);  // 16 MB
  _Float16* vt16   = (_Float16*)(ws + 134217728);  // 16 MB
  _Float16* ctx16  = (_Float16*)(ws + 150994944);  // 16 MB  (total 160 MB)

  cvt_f32_f16<<<4096, 256, 0, stream>>>(hs,   h16,    (size_t)4096 * 2048);
  cvt_f32_f16<<<4096, 256, 0, stream>>>(wqkv, wqkv16, (size_t)6144 * 2048);
  cvt_f32_f16<<<4096, 256, 0, stream>>>(wout, wout16, (size_t)2048 * 2048);

  // QKV projection: [4096,2048] x [6144,2048]^T -> [4096,6144] f16
  gemm_tn<true><<<dim3(48, 32), 256, 0, stream>>>(h16, wqkv16, qkv16,
                                                  4096, 6144, 2048);

  // RoPE + split into Q/K ([B,H,S,D]) and V^T ([B,H,D,S])
  rope_split<<<16384, 256, 0, stream>>>(qkv16, q16, k16, vt16);

  // causal flash attention -> ctx[b,s,e] f16
  flash_attn<<<dim3(16, 32), 256, 0, stream>>>(q16, k16, vt16, ctx16);

  // output projection: [4096,2048] x [2048,2048]^T -> d_out fp32
  gemm_tn<false><<<dim3(16, 32), 256, 0, stream>>>(ctx16, wout16, d_out,
                                                   4096, 2048, 2048);
}
